// GCN_Encoder_56281251447117
// MI455X (gfx1250) — matrix operand order repaired and checked by reading, not compiled, not run on hardware
//
#include <hip/hip_runtime.h>
#include <hip/hip_fp16.h>
#include <math.h>

typedef __attribute__((ext_vector_type(16))) _Float16 v16h;
typedef __attribute__((ext_vector_type(8)))  float    v8f;

#define EMBED  128
#define HIDDEN 256

// ---------------- graph scalar pipeline ----------------

// deg starts at 1.0 (self-loop), tacc at 0
__global__ void gcn_init_kernel(float* __restrict__ deg, float* __restrict__ tacc, int n) {
    int i = blockIdx.x * blockDim.x + threadIdx.x;
    if (i < n) { deg[i] = 1.0f; tacc[i] = 0.0f; }
}

// deg[dst] += 1 over real edges
__global__ void gcn_deg_kernel(const int* __restrict__ dst, float* __restrict__ deg, int E) {
    int e = blockIdx.x * blockDim.x + threadIdx.x;
    if (e < E) atomicAdd(&deg[dst[e]], 1.0f);
}

// deg -> dis = rsqrt(deg)   (deg >= 1 always, self-loop)
__global__ void gcn_rsq_kernel(float* __restrict__ deg, int n) {
    int i = blockIdx.x * blockDim.x + threadIdx.x;
    if (i < n) deg[i] = rsqrtf(deg[i]);
}

// tacc[dst] += dis[src] over real edges
__global__ void gcn_tacc_kernel(const int* __restrict__ src, const int* __restrict__ dst,
                                const float* __restrict__ dis, float* __restrict__ tacc, int E) {
    int e = blockIdx.x * blockDim.x + threadIdx.x;
    if (e < E) atomicAdd(&tacc[dst[e]], dis[src[e]]);
}

// s[n] = dis[n] * (tacc[n] + dis[n])   (self-loop message = dis^2), stored in tacc
__global__ void gcn_s_kernel(const float* __restrict__ dis, float* __restrict__ tacc, int n) {
    int i = blockIdx.x * blockDim.x + threadIdx.x;
    if (i < n) { float d = dis[i]; tacc[i] = d * (tacc[i] + d); }
}

// ---------------- W2 fragment pre-pack (f32 row-major -> wave32 WMMA B-fragment f16) ----------------
// B fragment layout used here (contiguous k, same order as A; 32x16, f16):
//   lane l: column n = l&15, element e (0..15): k = (l>=16 ? 16 : 0) + e
// Packed index: ((kk*8 + nt)*32 + lane)*16 + e  with K = kk*32 + ..., col = nt*16 + (l&15)
__global__ void gcn_pack_w2_kernel(const float* __restrict__ W2, _Float16* __restrict__ W2p) {
    int t = blockIdx.x * blockDim.x + threadIdx.x;   // 32768 threads total
    int e    = t & 15;
    int lane = (t >> 4) & 31;
    int nt   = (t >> 9) & 7;
    int kk   = (t >> 12) & 7;
    int K    = kk * 32 + ((lane >> 4) << 4) + e;
    int col  = nt * 16 + (lane & 15);
    W2p[t] = (_Float16)W2[K * EMBED + col];
}

// ---------------- fused rank-1 + ReLU + GEMM (WMMA) + bias + sigmoid + transpose ----------------
// One wave handles a 16-node tile; 8 waves/block share the 64 KB packed W2 in LDS.
// 64x v_wmma_f32_16x16x32_f16 per wave, B fragments via ds_load_b128.
__global__ __launch_bounds__(256) void gcn_wmma_kernel(
    const float* __restrict__ W1, const float* __restrict__ b1,
    const _Float16* __restrict__ W2p, const float* __restrict__ b2,
    const float* __restrict__ s, float* __restrict__ out, int n) {

    __shared__ _Float16 w2s[HIDDEN * EMBED];          // 64 KB of 320 KB WGP LDS

    // Cooperative fill: 4096 x b128 (global_load_b128 -> ds_store_b128)
    {
        const float4* gsrc = (const float4*)W2p;
        float4*       ldst = (float4*)w2s;
        #pragma unroll
        for (int i = 0; i < 16; ++i)
            ldst[i * 256 + threadIdx.x] = gsrc[i * 256 + threadIdx.x];
    }
    __syncthreads();

    const int lane = threadIdx.x & 31;
    const int wave = threadIdx.x >> 5;
    const int base = (blockIdx.x * (blockDim.x >> 5) + wave) * 16;
    if (base >= n) return;                    // whole-wave uniform exit (after barrier)

    const int m   = lane & 15;                // A row / D column-lane id
    const int hi  = lane >> 4;                // upper half-wave flag
    const int nd  = base + m;
    const float sval = (nd < n) ? s[nd] : 0.0f;
    const int kbase_a = hi * 16;              // A fragment K base per half-wave

    v8f acc[8] = {};                          // 8 embed tiles of 16, f32 accumulators

    #pragma unroll
    for (int kk = 0; kk < 8; ++kk) {
        // Build A fragment in-register: relu(W1[k]*s + b1[k]) as f16
        v16h a;
        #pragma unroll
        for (int e = 0; e < 16; ++e) {
            int k = kk * 32 + kbase_a + e;
            float h = fmaxf(fmaf(W1[k], sval, b1[k]), 0.0f);
            a[e] = (_Float16)h;
        }
        // B fragments from LDS: contiguous 32B per lane (2x ds_load_b128)
        const v16h* bp = (const v16h*)(w2s + kk * 4096 + lane * 16);
        #pragma unroll
        for (int nt = 0; nt < 8; ++nt) {
            v16h b = bp[nt * 32];             // nt*512 halves stride
            acc[nt] = __builtin_amdgcn_wmma_f32_16x16x32_f16(
                false, a, false, b, (short)0, acc[nt], false, false);
        }
    }

    // Epilogue: D(m_cd, n_cd) -> out[embed * n + node] with bias + sigmoid
    const int ncol = lane & 15;
    const int moff = hi * 8;
    #pragma unroll
    for (int nt = 0; nt < 8; ++nt) {
        int   oe = nt * 16 + ncol;
        float bb = b2[oe];
        #pragma unroll
        for (int r = 0; r < 8; ++r) {
            int node = base + moff + r;
            if (node < n) {
                float v = acc[nt][r] + bb;
                out[(size_t)oe * (size_t)n + node] = 1.0f / (1.0f + __expf(-v));
            }
        }
    }
}

extern "C" void kernel_launch(void* const* d_in, const int* in_sizes, int n_in,
                              void* d_out, int out_size, void* d_ws, size_t ws_size,
                              hipStream_t stream) {
    const int*   edge = (const int*)d_in[0];     // [2, E] row-major int32
    const float* W1   = (const float*)d_in[1];   // [1, 256]
    const float* b1   = (const float*)d_in[2];   // [256]
    const float* W2   = (const float*)d_in[3];   // [256, 128]
    const float* b2   = (const float*)d_in[4];   // [128]
    float*       out  = (float*)d_out;           // [128, N]

    const int E = in_sizes[0] / 2;
    const int n = out_size / EMBED;              // N nodes

    // workspace layout (32B-aligned slices): deg/dis [n] | tacc/s [n] | W2p [256*128 f16]
    float*    deg  = (float*)d_ws;
    float*    tacc = deg + n;
    _Float16* W2p  = (_Float16*)(tacc + n);

    const int* src = edge;
    const int* dst = edge + E;

    const int TB = 256;
    gcn_init_kernel<<<(n + TB - 1) / TB, TB, 0, stream>>>(deg, tacc, n);
    gcn_pack_w2_kernel<<<(HIDDEN * EMBED) / TB, TB, 0, stream>>>(W2, W2p);
    gcn_deg_kernel<<<(E + TB - 1) / TB, TB, 0, stream>>>(dst, deg, E);
    gcn_rsq_kernel<<<(n + TB - 1) / TB, TB, 0, stream>>>(deg, n);
    gcn_tacc_kernel<<<(E + TB - 1) / TB, TB, 0, stream>>>(src, dst, deg, tacc, E);
    gcn_s_kernel<<<(n + TB - 1) / TB, TB, 0, stream>>>(deg, tacc, n);

    const int waves_per_block = 8;               // 256 threads = 8 wave32
    const int tiles  = (n + 15) / 16;            // 6250 node tiles
    const int blocks = (tiles + waves_per_block - 1) / waves_per_block;
    gcn_wmma_kernel<<<blocks, 32 * waves_per_block, 0, stream>>>(W1, b1, W2p, b2, tacc, out, n);
}
